// Parser_44203803411083
// MI455X (gfx1250) — matrix operand order repaired and checked by reading, not compiled, not run on hardware
//
#include <hip/hip_runtime.h>

typedef __bf16 bf16;
typedef __attribute__((ext_vector_type(16))) __bf16 v16bf;
typedef __attribute__((ext_vector_type(8)))  float  v8f;

#define SEQ     384
#define NSENT   386
#define MPAD    400
#define OPAD    416
#define HC      256
#define HW      512
#define GC      1024   /* 4*HC */
#define GW      2048   /* 4*HW */
#define DW      512    /* 2*HC */
#define NCLS    128
#define CHARLEN 20

__device__ __forceinline__ float sigmoidf_(float x){ return 1.0f/(1.0f + __expf(-x)); }

__device__ __forceinline__ v8f wmma_bf(v16bf a, v16bf b, v8f c){
  // D(f32 16x16) = A(bf16 16x32) * B(bf16 32x16) + C
  return __builtin_amdgcn_wmma_f32_16x16x32_bf16(false, a, false, b, (short)0, c, false, false);
}

// A-matrix (16x32 bf16) register layout per ISA 7.12.2:
// lane m=l&15, half h=l>>4; VGPR v: k = (v>>2)*16 + h*8 + (v&3)*2 (pair)
__device__ __forceinline__ v16bf load_a_tile(const bf16* base, int ld, int k0, int lane){
  const int m = lane & 15, h = lane >> 4;
  const bf16* p = base + (size_t)m*ld + k0;
  v16bf a;
  #pragma unroll
  for (int v=0; v<8; ++v){
    const int kk = (v>>2)*16 + h*8 + (v&3)*2;
    a[2*v]   = p[kk];
    a[2*v+1] = p[kk+1];
  }
  return a;
}

// B-matrix (32x16 bf16), B[k][n] = W[n][k] for row-major W[N,K]:
// lane n=l&15, half h=l>>4; VGPR v: k = (v>>2)*16 + h*8 + (v&3)*2 (pair)
__device__ __forceinline__ v16bf load_b_tile(const bf16* W, int ld, int n0, int k0, int lane){
  const int n = lane & 15, h = lane >> 4;
  const bf16* p = W + (size_t)(n0+n)*ld + k0;
  v16bf b;
  #pragma unroll
  for (int v=0; v<8; ++v){ const int kk = (v>>2)*16 + h*8 + (v&3)*2; b[2*v] = p[kk]; b[2*v+1] = p[kk+1]; }
  return b;
}

// ------------------------------------------------------------------ utilities
__global__ void zero32_kernel(unsigned int* __restrict__ p, int n){
  int i = blockIdx.x*256 + threadIdx.x;
  if (i < n) p[i] = 0u;
}
__global__ void f32_to_bf16_kernel(const float* __restrict__ s, bf16* __restrict__ d, int n){
  int i = blockIdx.x*256 + threadIdx.x;
  if (i < n) d[i] = (bf16)s[i];
}
__global__ void split_wlin_kernel(const float* __restrict__ W, bf16* __restrict__ Wf, bf16* __restrict__ Wb){
  int i = blockIdx.x*256 + threadIdx.x;       // NCLS*HW
  int r = i >> 9, k = i & 511;
  Wf[i] = (bf16)W[(size_t)r*1024 + k];
  Wb[i] = (bf16)W[(size_t)r*1024 + 512 + k];
}
// embW[v][c] = sum_e emb[v][e]*Wih[c][e] + b[c]   -> gx becomes a pure gather
__global__ void embw_kernel(const float* __restrict__ emb, const float* __restrict__ Wih,
                            const float* __restrict__ bias, float* __restrict__ embW){
  int i = blockIdx.x*256 + threadIdx.x;       // 128*GC
  int v = i >> 10, c = i & (GC-1);
  const float* e = emb + v*64;
  const float* w = Wih + (size_t)c*64;
  float acc = bias[c];
  #pragma unroll 8
  for (int k=0; k<64; ++k) acc += e[k]*w[k];
  embW[i] = acc;
}

// ------------------------------------------------------- char-level LSTMs (WMMA)
// grid (24, 2), 512 threads = 16 waves. 16 words per block; blockIdx.y selects
// prefix/suffix. Per step: g[16,1024] = gather(embW) + h @ Whh^T via
// v_wmma_f32_16x16x32_bf16. Wave w owns hidden block [w*16, w*16+16) across all
// 4 gates, so gate fusion is wave-local: acc[4] (32 VGPRs) keeps pressure low.
// The memory clobber stops the compiler from hoisting the t-invariant Whh tiles
// out of the recurrence (which previously spilled ~512 VGPRs to scratch).
__global__ void __launch_bounds__(512) char_lstm_kernel(
    const int* __restrict__ prefix, const int* __restrict__ suffix,
    const float* __restrict__ embW_p, const float* __restrict__ embW_s,
    const bf16* __restrict__ WhhP, const bf16* __restrict__ WhhS,
    bf16* __restrict__ sent)
{
  __shared__ bf16 h_s[16*HC];
  const int type = blockIdx.y;
  const int* chars      = type ? suffix : prefix;
  const float* embW     = type ? embW_s : embW_p;
  const bf16* Whh       = type ? WhhS   : WhhP;
  const int wt  = blockIdx.x;
  const int tid = threadIdx.x;
  const int lane = tid & 31, w = tid >> 5;    // w = 0..15: hidden block bt = w
  const int h = lane >> 4, nn = lane & 15;

  for (int i = tid; i < 16*HC; i += 512) h_s[i] = (bf16)0.0f;
  float creg[8];
  #pragma unroll
  for (int v=0; v<8; ++v) creg[v] = 0.0f;
  __syncthreads();

  #pragma unroll 1
  for (int t=0; t<CHARLEN; ++t){
    asm volatile("" ::: "memory");   // keep Whh tile loads inside the t loop

    v8f acc[4];
    #pragma unroll
    for (int G=0; G<4; ++G)
      #pragma unroll
      for (int v=0; v<8; ++v) acc[G][v] = 0.0f;

    #pragma unroll 1
    for (int kt=0; kt<HC/32; ++kt){
      v16bf a = load_a_tile(h_s, HC, kt*32, lane);
      #pragma unroll
      for (int G=0; G<4; ++G){
        v16bf b = load_b_tile(Whh, HC, G*HC + w*16, kt*32, lane);
        acc[G] = wmma_bf(a, b, acc[G]);
      }
    }
    __syncthreads();              // all reads of h_s complete before rewrite

    const int hidx = w*16 + nn;
    #pragma unroll
    for (int v=0; v<8; ++v){
      const int m = v + h*8;
      const int word = wt*16 + m;
      const int ch = chars[word*CHARLEN + t];
      const float* er = embW + (size_t)ch*GC;
      float gi = acc[0][v] + er[hidx];
      float gf = acc[1][v] + er[HC   + hidx];
      float gg = acc[2][v] + er[2*HC + hidx];
      float go = acc[3][v] + er[3*HC + hidx];
      float cc = sigmoidf_(gf)*creg[v] + sigmoidf_(gi)*tanhf(gg);
      creg[v] = cc;
      h_s[m*HC + hidx] = (bf16)(sigmoidf_(go)*tanhf(cc));
    }
    __syncthreads();
  }
  // final h -> sentence buffer row (1+word), half selected by type
  for (int i = tid; i < 16*HC; i += 512){
    const int m = i >> 8, hidx = i & (HC-1);
    sent[(size_t)(1 + wt*16 + m)*DW + type*HC + hidx] = h_s[i];
  }
}

// ----------------------------------------------- gx = sent @ Wih^T + b  (WMMA)
// grid (25, 16): M-tile x 128-column group; 8 waves x 1 n-tile.
__global__ void __launch_bounds__(256) gx_gemm_kernel(
    const bf16* __restrict__ A, const bf16* __restrict__ W,
    const float* __restrict__ bias, float* __restrict__ out)
{
  const int lane = threadIdx.x & 31, w = threadIdx.x >> 5;
  const int mt = blockIdx.x;
  const int nt = blockIdx.y*8 + w;
  const bf16* atile = A + (size_t)mt*16*DW;
  v8f acc;
  #pragma unroll
  for (int v=0; v<8; ++v) acc[v] = 0.0f;
  for (int kt=0; kt<DW/32; ++kt){
    if (kt+1 < DW/32)
      __builtin_prefetch(W + (size_t)(nt*16 + (lane&15))*DW + (kt+1)*32, 0, 1);
    v16bf a = load_a_tile(atile, DW, kt*32, lane);
    v16bf b = load_b_tile(W, DW, nt*16, kt*32, lane);
    acc = wmma_bf(a, b, acc);
  }
  const int n = nt*16 + (lane&15), h = lane >> 4;
  const float bv = bias[n];
  #pragma unroll
  for (int v=0; v<8; ++v) out[(size_t)(mt*16 + h*8 + v)*GW + n] = acc[v] + bv;
}

// --------------------------------------- batch-1 bidirectional sentence LSTM
// grid 16: blocks 0-7 = forward, 8-15 = backward; each block owns 64 hidden
// positions (256 gate rows). Per step: gate dots from L2-resident bf16 Whh,
// ping-pong h buffer, agent-scope atomic split barrier across the 8 blocks.
__global__ void __launch_bounds__(256) sent_lstm_kernel(
    const float* __restrict__ gxf, const float* __restrict__ gxb,
    const bf16* __restrict__ WhhF, const bf16* __restrict__ WhhB,
    float* __restrict__ hbuf, bf16* __restrict__ outf, bf16* __restrict__ outb,
    unsigned int* __restrict__ cnt)
{
  __shared__ float gd[256];
  const int dir = blockIdx.x >> 3;
  const int g   = blockIdx.x & 7;
  const float* gx = dir ? gxb  : gxf;
  const bf16*  W  = dir ? WhhB : WhhF;
  bf16* outp      = dir ? outb : outf;
  const int tid  = threadIdx.x;
  const int gate = tid >> 6;
  const int grow = gate*HW + g*64 + (tid & 63);
  const bf16* wr = W + (size_t)grow*HW;
  float* hb = hbuf + dir*2*HW;
  unsigned int* c = cnt + dir*32;
  float creg = 0.0f;

  #pragma unroll 1
  for (int s=0; s<NSENT; ++s){
    if (tid == 0){
      while (__hip_atomic_load(c, __ATOMIC_ACQUIRE, __HIP_MEMORY_SCOPE_AGENT)
             < (unsigned)(8*s))
        __builtin_amdgcn_s_sleep(1);
    }
    __syncthreads();
    const int t = dir ? (NSENT-1 - s) : s;
    const float* h = hb + (s & 1)*HW;
    float acc = gx[(size_t)t*GW + grow];
    #pragma unroll 8
    for (int k=0; k<HW; ++k) acc += (float)wr[k] * h[k];
    gd[tid] = acc;
    __syncthreads();
    if (tid < 64){
      float gi = gd[tid], gf = gd[64+tid], gg = gd[128+tid], go = gd[192+tid];
      creg = sigmoidf_(gf)*creg + sigmoidf_(gi)*tanhf(gg);
      float hh = sigmoidf_(go)*tanhf(creg);
      const int pos = g*64 + tid;
      hb[((s+1)&1)*HW + pos] = hh;          // h for next step (other buffer)
      outp[(size_t)t*HW + pos] = (bf16)hh;  // out_f[t] / out_b_unreversed[t]
      __threadfence();
    }
    __syncthreads();
    if (tid == 0)
      __hip_atomic_fetch_add(c, 1u, __ATOMIC_RELEASE, __HIP_MEMORY_SCOPE_AGENT);
  }
}

// --------------------------------------- P = fwd @ Wf^T,  Q = bwd @ Wb^T  (WMMA)
// grid 25: one M-tile per block; 8 waves cover all 128 output columns.
__global__ void __launch_bounds__(256) pq_gemm_kernel(
    const bf16* __restrict__ A, const bf16* __restrict__ W, float* __restrict__ out)
{
  const int lane = threadIdx.x & 31, w = threadIdx.x >> 5;
  const int mt = blockIdx.x;
  const bf16* atile = A + (size_t)mt*16*HW;
  v8f acc;
  #pragma unroll
  for (int v=0; v<8; ++v) acc[v] = 0.0f;
  for (int kt=0; kt<HW/32; ++kt){
    v16bf a = load_a_tile(atile, HW, kt*32, lane);
    v16bf b = load_b_tile(W, HW, w*16, kt*32, lane);
    acc = wmma_bf(a, b, acc);
  }
  const int n = w*16 + (lane&15), h = lane >> 4;
  #pragma unroll
  for (int v=0; v<8; ++v) out[(size_t)(mt*16 + h*8 + v)*NCLS + n] = acc[v];
}

// ------------- out(i,j) = P[j]-P[i] + Q[i]-Q[j] + b  over all 73920 spans
__global__ void __launch_bounds__(128) span_kernel(
    const float* __restrict__ P, const float* __restrict__ Q,
    const float* __restrict__ bias, float* __restrict__ out)
{
  const int i = blockIdx.x;                 // 0..383 (start)
  const int c = threadIdx.x;                // 0..127
  const int j0 = i + 1 + blockIdx.y*128;
  const float Ai = P[(size_t)i*NCLS + c];
  const float Bi = Q[(size_t)i*NCLS + c];
  const float bc = bias[c];
  // idx(i,j) = 384*i - i*(i-1)/2 + (j - i - 1)
  const long base = (long)384*i - (long)i*(i-1)/2 - i - 1;
  for (int jj=0; jj<128; ++jj){
    const int j = j0 + jj;
    if (j > 384) break;
    const long idx = base + j;
    out[idx*NCLS + c] = P[(size_t)j*NCLS + c] - Ai + Bi - Q[(size_t)j*NCLS + c] + bc;
  }
}

// =============================================================== host driver
extern "C" void kernel_launch(void* const* d_in, const int* in_sizes, int n_in,
                              void* d_out, int out_size, void* d_ws, size_t ws_size,
                              hipStream_t stream)
{
  (void)in_sizes; (void)n_in; (void)out_size; (void)ws_size;
  const int*   prefix = (const int*)  d_in[0];
  const int*   suffix = (const int*)  d_in[1];
  const float* emb    = (const float*)d_in[2];
  const float* Wih_p  = (const float*)d_in[3];
  const float* Whh_p  = (const float*)d_in[4];
  const float* b_p    = (const float*)d_in[5];
  const float* Wih_s  = (const float*)d_in[6];
  const float* Whh_s  = (const float*)d_in[7];
  const float* b_s    = (const float*)d_in[8];
  const float* Wih_f  = (const float*)d_in[9];
  const float* Whh_f  = (const float*)d_in[10];
  const float* b_f    = (const float*)d_in[11];
  const float* Wih_b  = (const float*)d_in[12];
  const float* Whh_b  = (const float*)d_in[13];
  const float* b_b    = (const float*)d_in[14];
  const float* W_lin  = (const float*)d_in[15];
  const float* b_lin  = (const float*)d_in[16];
  float* out = (float*)d_out;

  char* base = (char*)d_ws; size_t off = 0;
  auto alloc = [&](size_t b)->char* {
    char* p = base + off; off = (off + b + 255) & ~(size_t)255; return p;
  };
  float* embWp = (float*)alloc((size_t)128*GC*4);
  float* embWs = (float*)alloc((size_t)128*GC*4);
  bf16*  WhhPb = (bf16*) alloc((size_t)GC*HC*2);
  bf16*  WhhSb = (bf16*) alloc((size_t)GC*HC*2);
  bf16*  WihFb = (bf16*) alloc((size_t)GW*DW*2);
  bf16*  WihBb = (bf16*) alloc((size_t)GW*DW*2);
  bf16*  WhhFb = (bf16*) alloc((size_t)GW*HW*2);
  bf16*  WhhBb = (bf16*) alloc((size_t)GW*HW*2);
  bf16*  Wfb   = (bf16*) alloc((size_t)NCLS*HW*2);
  bf16*  Wbb   = (bf16*) alloc((size_t)NCLS*HW*2);
  bf16*  sentb = (bf16*) alloc((size_t)MPAD*DW*2);
  float* gxf   = (float*)alloc((size_t)MPAD*GW*4);
  float* gxb   = (float*)alloc((size_t)MPAD*GW*4);
  bf16*  outfb = (bf16*) alloc((size_t)OPAD*HW*2);
  bf16*  outbb = (bf16*) alloc((size_t)OPAD*HW*2);
  float* P     = (float*)alloc((size_t)MPAD*NCLS*4);
  float* Q     = (float*)alloc((size_t)MPAD*NCLS*4);
  float* hbuf  = (float*)alloc((size_t)2*2*HW*4);
  unsigned int* cnt = (unsigned int*)alloc(64*4);

  auto nb = [](int n){ return (n + 255) / 256; };

  // init (re-zeroed every call -> deterministic under graph replay)
  zero32_kernel<<<nb(MPAD*DW/2),   256, 0, stream>>>((unsigned int*)sentb, MPAD*DW/2);
  zero32_kernel<<<nb(OPAD*HW/2),   256, 0, stream>>>((unsigned int*)outfb, OPAD*HW/2);
  zero32_kernel<<<nb(OPAD*HW/2),   256, 0, stream>>>((unsigned int*)outbb, OPAD*HW/2);
  zero32_kernel<<<nb(2*2*HW),      256, 0, stream>>>((unsigned int*)hbuf,  2*2*HW);
  zero32_kernel<<<1,               256, 0, stream>>>(cnt, 64);

  // weight conversions to bf16
  f32_to_bf16_kernel<<<nb(GC*HC), 256, 0, stream>>>(Whh_p, WhhPb, GC*HC);
  f32_to_bf16_kernel<<<nb(GC*HC), 256, 0, stream>>>(Whh_s, WhhSb, GC*HC);
  f32_to_bf16_kernel<<<nb(GW*DW), 256, 0, stream>>>(Wih_f, WihFb, GW*DW);
  f32_to_bf16_kernel<<<nb(GW*DW), 256, 0, stream>>>(Wih_b, WihBb, GW*DW);
  f32_to_bf16_kernel<<<nb(GW*HW), 256, 0, stream>>>(Whh_f, WhhFb, GW*HW);
  f32_to_bf16_kernel<<<nb(GW*HW), 256, 0, stream>>>(Whh_b, WhhBb, GW*HW);
  split_wlin_kernel<<<nb(NCLS*HW), 256, 0, stream>>>(W_lin, Wfb, Wbb);

  // embW = emb @ Wih^T + b (folds the char input projection into a gather table)
  embw_kernel<<<nb(128*GC), 256, 0, stream>>>(emb, Wih_p, b_p, embWp);
  embw_kernel<<<nb(128*GC), 256, 0, stream>>>(emb, Wih_s, b_s, embWs);

  // char LSTMs -> sentence buffer rows 1..384 (rows 0 and 385 stay zero)
  char_lstm_kernel<<<dim3(SEQ/16, 2), 512, 0, stream>>>(
      prefix, suffix, embWp, embWs, WhhPb, WhhSb, sentb);

  // gx for both sentence directions
  gx_gemm_kernel<<<dim3(MPAD/16, GW/128), 256, 0, stream>>>(sentb, WihFb, b_f, gxf);
  gx_gemm_kernel<<<dim3(MPAD/16, GW/128), 256, 0, stream>>>(sentb, WihBb, b_b, gxb);

  // sequential bidirectional sentence LSTM (16 cooperating workgroups)
  sent_lstm_kernel<<<16, 256, 0, stream>>>(gxf, gxb, WhhFb, WhhBb, hbuf, outfb, outbb, cnt);

  // P = out_f[0:385] @ Wf^T ; Q = out_b[1:386] @ Wb^T
  pq_gemm_kernel<<<MPAD/16, 256, 0, stream>>>(outfb,      Wfb, P);
  pq_gemm_kernel<<<MPAD/16, 256, 0, stream>>>(outbb + HW, Wbb, Q);

  // span assembly: out(i,j) = P[j]-P[i]+Q[i]-Q[j]+b
  span_kernel<<<dim3(SEQ, 3), 128, 0, stream>>>(P, Q, b_lin, out);
}
